// CapsLayer_27401891349222
// MI455X (gfx1250) — matrix operand order repaired and checked by reading, not compiled, not run on hardware
//
#include <hip/hip_runtime.h>
#include <hip/hip_bf16.h>

// B=1024, NUM_CAPS=2, IN_CAPS=512, IN_CH=256, OUT_CH=64, ITER=3
//
// Phase 1: u_hat[b,n,cap,o] = sum_i W[n,cap,o,i] * x[b,cap,i]
//   -> per (cap): [1024x256] x [256x64] x {n=0,1} GEMM via v_wmma_f32_16x16x32_bf16
//   -> u_hat stored bf16 in d_ws (134 MB: fits MI455X 192MB L2)
//   -> next X chunk prefetched (global_prefetch_b8) while WMMAs run
// Phase 2: dynamic routing, one workgroup per sample b; u_hat[b] (128KB bf16)
//   staged once into CDNA5's 320KB LDS, all 3 iterations run on-chip.

typedef __attribute__((ext_vector_type(16))) __bf16 v16bf;
typedef __attribute__((ext_vector_type(8)))  float  v8f;

union BFFrag {
    unsigned int u[8];
    v16bf v;
};

__device__ __forceinline__ unsigned short f32_to_bf16(float f) {
    unsigned int u = __float_as_uint(f);
    unsigned int r = u + 0x7FFFu + ((u >> 16) & 1u);   // round-to-nearest-even
    return (unsigned short)(r >> 16);
}

__device__ __forceinline__ unsigned int pack_bf16x2(float lo, float hi) {
    return (unsigned int)f32_to_bf16(lo) | ((unsigned int)f32_to_bf16(hi) << 16);
}

__device__ __forceinline__ float bf16_bits_to_f32(unsigned int hbits) {
    return __uint_as_float(hbits << 16);
}

// ---------------------------------------------------------------------------
// Phase 1: u_hat GEMM. grid = 512 (one block per input capsule), 256 threads.
// ---------------------------------------------------------------------------
__global__ __launch_bounds__(256) void caps_uhat_gemm(
    const float* __restrict__ X,            // (1024, 512, 256) f32
    const float* __restrict__ W,            // (2, 512, 64, 256) f32
    unsigned short* __restrict__ uhat)      // (1024, 2, 512, 64) bf16
{
    constexpr int STR = 258;                // ushort row stride (129 dwords: conflict-free)
    __shared__ unsigned short Ws[128 * STR];   // rows: n*64+o, cols: i   (66048 B)
    __shared__ unsigned short Xs[128 * STR];   // rows: m (chunk), cols: i (66048 B)

    const int cap  = blockIdx.x;
    const int t    = threadIdx.x;
    const int lane = t & 31;
    const int wave = t >> 5;
    const int half = lane >> 4;          // lane group (K-half select)
    const int ml   = lane & 15;          // M / N index within tile

    // ---- Stage W[0..1][cap] -> bf16 LDS (32768 floats, 32 float4/thread) ----
    for (int c4 = t; c4 < 8192; c4 += 256) {
        int flat = c4 << 2;
        int row  = flat >> 8;                       // n*64 + o, 0..127
        int i    = flat & 255;
        size_t g = (((size_t)(row >> 6) * 512 + cap) * 64 + (row & 63)) * 256 + i;
        float4 wv = *(const float4*)(W + g);
        unsigned int* dst = (unsigned int*)(Ws + row * STR + i);
        dst[0] = pack_bf16x2(wv.x, wv.y);
        dst[1] = pack_bf16x2(wv.z, wv.w);
    }

    for (int mc = 0; mc < 8; ++mc) {            // 8 chunks of 128 rows of B=1024
        // ---- Stage X chunk -> bf16 LDS ----
        for (int c4 = t; c4 < 8192; c4 += 256) {
            int r = c4 >> 6;
            int i = (c4 & 63) << 2;
            size_t g = ((size_t)(mc * 128 + r) * 512 + cap) * 256 + i;
            float4 xv = *(const float4*)(X + g);
            unsigned int* dst = (unsigned int*)(Xs + r * STR + i);
            dst[0] = pack_bf16x2(xv.x, xv.y);
            dst[1] = pack_bf16x2(xv.z, xv.w);
        }
        __syncthreads();

        // ---- Prefetch next X chunk into cache while WMMAs run ----
        if (mc < 7) {
            const float* nxt = X + ((size_t)((mc + 1) * 128) * 512 + cap) * 256;
            for (int l = t; l < 1024; l += 256) {          // 1024 x 128B lines
                int r   = l >> 3;
                int off = (l & 7) * 32;                    // 32 floats = 128B
                __builtin_prefetch(nxt + (size_t)r * 512 * 256 + off, 0, 1);
            }
        }

        // Wave owns M-tile `wave`; computes 8 output tiles: (n in 2) x (ntile in 4)
        v8f acc[8];
#pragma unroll
        for (int tl = 0; tl < 8; ++tl)
#pragma unroll
            for (int e = 0; e < 8; ++e) acc[tl][e] = 0.0f;

        const int m = (wave << 4) + ml;     // A-matrix row held by this lane

#pragma unroll 1
        for (int k0 = 0; k0 < 256; k0 += 32) {
            // A fragment, 16x32 bf16 (ISA 7.12.2 layout)
            BFFrag a;
#pragma unroll
            for (int j = 0; j < 8; ++j) {
                int koff = (j < 4) ? (half * 8 + 2 * j)
                                   : (16 + half * 8 + 2 * (j - 4));
                a.u[j] = *(const unsigned int*)(Xs + m * STR + k0 + koff);
            }
#pragma unroll
            for (int tl = 0; tl < 8; ++tl) {
                int n  = tl >> 2;
                int nt = tl & 3;
                int wrow = n * 64 + nt * 16 + ml;   // B col (o) for this lane
                // B fragment, 32x16 bf16: same K order as A (ISA 7.12.2 layout)
                BFFrag bf;
#pragma unroll
                for (int j = 0; j < 8; ++j) {
                    int k = k0 + ((j < 4) ? (half * 8 + 2 * j) : (16 + half * 8 + 2 * (j - 4)));
                    bf.u[j] = *(const unsigned int*)(Ws + wrow * STR + k);
                }
                acc[tl] = __builtin_amdgcn_wmma_f32_16x16x32_bf16(
                    false, a.v, false, bf.v, (short)0, acc[tl], false, false);
            }
        }

        // ---- Store D tiles as bf16: D VGPR j = row (j + 8*half), col = ml ----
#pragma unroll
        for (int tl = 0; tl < 8; ++tl) {
            int n  = tl >> 2;
            int nt = tl & 3;
            int o  = nt * 16 + ml;
#pragma unroll
            for (int j = 0; j < 8; ++j) {
                int bb = mc * 128 + (wave << 4) + (half << 3) + j;
                size_t idx = (((size_t)bb * 2 + n) * 512 + cap) * 64 + o;
                uhat[idx] = f32_to_bf16(acc[tl][j]);
            }
        }
        __syncthreads();
    }
}

// ---------------------------------------------------------------------------
// Phase 2: routing. grid = 1024 (one block per sample), 256 threads.
// u_hat[b] (2*512*64 bf16 = 128KB) lives in LDS for all 3 iterations.
// ---------------------------------------------------------------------------
__global__ __launch_bounds__(256) void caps_route(
    const unsigned short* __restrict__ uhat,   // (1024, 2, 512, 64) bf16
    float* __restrict__ out)                   // (1024, 2, 64) f32
{
    constexpr int USTR = 68;                   // padded row stride (34 dwords)
    __shared__ unsigned short uh[1024 * USTR]; // rows: n*512+cap  (139264 B)
    __shared__ float blog[1024];               // routing logits b[n][cap]
    __shared__ float ps[256];                  // partial sums for s
    __shared__ float sarr[128];                // s[n][o]
    __shared__ float varr[128];                // v[n][o]
    __shared__ float scl[2];                   // squash scales

    const int b = blockIdx.x;
    const int t = threadIdx.x;

    // ---- Load u_hat[b] -> LDS (8192 uint4) ----
    const unsigned short* src = uhat + (size_t)b * 65536;
    for (int ch = t; ch < 8192; ch += 256) {
        int row  = ch >> 3;
        int part = ch & 7;
        uint4 vsrc = *(const uint4*)(src + ch * 8);
        uint2* d = (uint2*)(uh + row * USTR + part * 8);
        uint2 lo; lo.x = vsrc.x; lo.y = vsrc.y;
        uint2 hi; hi.x = vsrc.z; hi.y = vsrc.w;
        d[0] = lo;
        d[1] = hi;
    }
    for (int p = t; p < 1024; p += 256) blog[p] = 0.0f;
    __syncthreads();

    for (int iter = 0; iter < 3; ++iter) {
        // ---- s[n][o] = sum_cap softmax_n(b)[n][cap] * uh[n][cap][o] ----
        {
            int no = t & 127;          // n*64 + o
            int n  = no >> 6;
            int o  = no & 63;
            int h  = t >> 7;           // cap half
            float acc = 0.0f;
            int capBase = h << 8;
            for (int c = 0; c < 256; ++c) {
                int cap = capBase + c;
                float b0 = blog[cap];
                float b1 = blog[512 + cap];
                float mx = fmaxf(b0, b1);
                float e0 = __expf(b0 - mx);
                float e1 = __expf(b1 - mx);
                float cc = ((n == 0) ? e0 : e1) / (e0 + e1);
                float u  = bf16_bits_to_f32(
                    (unsigned int)uh[(n * 512 + cap) * USTR + o]);
                acc = fmaf(cc, u, acc);
            }
            ps[(h << 7) + no] = acc;
        }
        __syncthreads();
        if (t < 128) sarr[t] = ps[t] + ps[128 + t];
        __syncthreads();
        // ---- squash scale per n: |s|^2/(1+|s|^2)/|s| = sqrt(|s|^2)/(1+|s|^2) ----
        if (t < 2) {
            float sq = 0.0f;
            for (int o = 0; o < 64; ++o) {
                float s = sarr[(t << 6) + o];
                sq = fmaf(s, s, sq);
            }
            scl[t] = sqrtf(sq) / (1.0f + sq);
        }
        __syncthreads();
        if (t < 128) {
            float v = scl[t >> 6] * sarr[t];
            varr[t] = v;
            if (iter == 2) out[(size_t)b * 128 + t] = v;   // final output v
        }
        __syncthreads();
        // ---- b[n][cap] += <uh[n][cap][:], v[n][:]>  (skip after last v) ----
        if (iter < 2) {
            for (int jj = 0; jj < 4; ++jj) {
                int p = t + (jj << 8);           // n*512 + cap
                int n = p >> 9;
                const unsigned short* rowp = uh + p * USTR;
                const float* vp = varr + (n << 6);
                float dot = 0.0f;
#pragma unroll 8
                for (int oi = 0; oi < 32; ++oi) {
                    unsigned int u2 = *(const unsigned int*)(rowp + 2 * oi);
                    float ulo = __uint_as_float(u2 << 16);
                    float uhi = __uint_as_float(u2 & 0xFFFF0000u);
                    dot = fmaf(ulo, vp[2 * oi], dot);
                    dot = fmaf(uhi, vp[2 * oi + 1], dot);
                }
                blog[p] += dot;
            }
            __syncthreads();
        }
    }
}

// ---------------------------------------------------------------------------
extern "C" void kernel_launch(void* const* d_in, const int* in_sizes, int n_in,
                              void* d_out, int out_size, void* d_ws, size_t ws_size,
                              hipStream_t stream) {
    const float* x = (const float*)d_in[0];   // (1024, 512, 256)
    const float* W = (const float*)d_in[1];   // (1, 2, 512, 64, 256)
    float* out = (float*)d_out;               // (1024, 2, 64)
    unsigned short* uhat = (unsigned short*)d_ws;  // 1024*2*512*64 bf16 = 128 MB

    caps_uhat_gemm<<<512, 256, 0, stream>>>(x, W, uhat);
    caps_route<<<1024, 256, 0, stream>>>(uhat, out);
}
